// PointNetSetAbstraction_58892591563298
// MI455X (gfx1250) — compile-verified, matrix operand order checked by reading, not run on hardware
//
#include <hip/hip_runtime.h>
#include <hip/hip_bf16.h>

typedef __attribute__((ext_vector_type(16))) _Float16 v16h;
typedef __attribute__((ext_vector_type(8)))  float    v8f;

constexpr int Bn = 16, Np = 4096, Sc = 1024, Kn = 32, Kx2 = 64, Dn = 64;
constexpr int CIN = 67, PK1 = 96, H1 = 64, H2 = 64, H3 = 128;
constexpr int ROWS = Bn * Sc;                 // 16384 (b,s) rows
constexpr size_t MT = (size_t)ROWS * Kn;      // 524288 MLP rows
constexpr float EPSN = 1e-5f;

// ---------------- fragment-position swizzle (gfx1250 wave32 f16 A/B layout) --
// Element (mn, kk) of a 16x32 fragment lives at lane = mn + 16*h, half-index
// 2*v + p where: p = kk&1, q = kk>>1, h = (q>>2)&1, v = (q&3) | ((q>>3)<<2).
// (Inverse of: v<4 -> k = 2v+8h ; v>=4 -> k = 16+2(v-4)+8h.)
__device__ __forceinline__ int frag_off(int mn, int kk) {
  int p = kk & 1, q = kk >> 1;
  int h = (q >> 2) & 1;
  int v = (q & 3) | ((q >> 3) << 2);
  return ((mn + (h << 4)) << 4) + (v << 1) + p;   // lane*16 + 2v + p
}
__device__ __forceinline__ v16h load_frag(const _Float16* buf, int lane) {
  return *(const v16h*)(buf + lane * 16);         // 32B contiguous -> 2x ds_load_b128
}

// ---------------- 1) Farthest point sampling (1 block / batch) ---------------
__global__ void fps_kernel(const float* __restrict__ data, const int* __restrict__ initf,
                           float* __restrict__ cent, float* __restrict__ out_cent) {
  __shared__ float px[Np], py[Np], pz[Np];   // 48KB
  __shared__ float rv[32];
  __shared__ int   ri[32];
  __shared__ int   s_far;
  int b = blockIdx.x, tid = threadIdx.x;     // 1024 threads
  const float* dB = data + (size_t)b * Np * 3;
  for (int i = tid; i < Np; i += 1024) {
    px[i] = dB[i * 3 + 0]; py[i] = dB[i * 3 + 1]; pz[i] = dB[i * 3 + 2];
  }
  float dloc[4];
#pragma unroll
  for (int k = 0; k < 4; ++k) dloc[k] = 1e10f;
  if (tid == 0) s_far = initf[b];
  __syncthreads();
  int lane = tid & 31, wv = tid >> 5;
  for (int s = 0; s < Sc; ++s) {
    int far = s_far;
    float cx = px[far], cy = py[far], cz = pz[far];
    if (tid == 0) {
      size_t o = ((size_t)b * Sc + s) * 3;
      cent[o] = cx; cent[o + 1] = cy; cent[o + 2] = cz;
      out_cent[o] = cx; out_cent[o + 1] = cy; out_cent[o + 2] = cz;
    }
    float bv = -1.0f; int bi = 0x7fffffff;
#pragma unroll
    for (int k = 0; k < 4; ++k) {
      int i = tid + k * 1024;
      float dx = px[i] - cx, dy = py[i] - cy, dz = pz[i] - cz;
      float d = dx * dx + dy * dy + dz * dz;
      if (d < dloc[k]) dloc[k] = d;
      float dd = dloc[k];
      if (dd > bv || (dd == bv && i < bi)) { bv = dd; bi = i; }
    }
    for (int off = 16; off > 0; off >>= 1) {
      float ov = __shfl_xor(bv, off, 32);
      int   oi = __shfl_xor(bi, off, 32);
      if (ov > bv || (ov == bv && oi < bi)) { bv = ov; bi = oi; }
    }
    if (lane == 0) { rv[wv] = bv; ri[wv] = bi; }
    __syncthreads();
    if (wv == 0) {
      float v = rv[lane]; int idx = ri[lane];
      for (int off = 16; off > 0; off >>= 1) {
        float ov = __shfl_xor(v, off, 32);
        int   oi = __shfl_xor(idx, off, 32);
        if (ov > v || (ov == v && oi < idx)) { v = ov; idx = oi; }
      }
      if (lane == 0) s_far = idx;
    }
    __syncthreads();
  }
}

// ---------------- 2) Neighbor selection + softmax weights (1 block / (b,s)) --
__global__ void neighbor_kernel(const float* __restrict__ data, const float* __restrict__ cent,
                                float* __restrict__ snw, int* __restrict__ tki_out,
                                float* __restrict__ zw_out) {
  __shared__ float dist[Np];                 // 16KB
  __shared__ float sx[Kx2], sy[Kx2], sz[Kx2], score[Kx2];
  __shared__ int   stki[Kn];
  __shared__ float sw[Kn];
  __shared__ float rv[8];
  __shared__ int   ri[8];
  __shared__ float s_inv;
  int row = blockIdx.x;                      // b*Sc + s
  int b = row >> 10;
  int tid = threadIdx.x;                     // 256 threads
  const float* dB = data + (size_t)b * Np * 3;
  float cx = cent[row * 3], cy = cent[row * 3 + 1], cz = cent[row * 3 + 2];
  float cs = cx * cx + cy * cy + cz * cz;
  for (int i = tid; i < Np; i += 256) {
    float x = dB[i * 3], y = dB[i * 3 + 1], z = dB[i * 3 + 2];
    dist[i] = -2.0f * (x * cx + y * cy + z * cz) + (x * x + y * y + z * z) + cs;
  }
  __syncthreads();
  int lane = tid & 31, wv = tid >> 5;
  for (int j = 0; j < Kx2; ++j) {            // top-64 LARGEST (replicates reference)
    float bv = -1e30f; int bi = 0x7fffffff;
    for (int i = tid; i < Np; i += 256) {
      float d = dist[i];
      if (d > bv || (d == bv && i < bi)) { bv = d; bi = i; }
    }
    for (int off = 16; off > 0; off >>= 1) {
      float ov = __shfl_xor(bv, off, 32);
      int   oi = __shfl_xor(bi, off, 32);
      if (ov > bv || (ov == bv && oi < bi)) { bv = ov; bi = oi; }
    }
    if (lane == 0) { rv[wv] = bv; ri[wv] = bi; }
    __syncthreads();
    if (tid == 0) {
      float v = rv[0]; int idx = ri[0];
      for (int w2 = 1; w2 < 8; ++w2)
        if (rv[w2] > v || (rv[w2] == v && ri[w2] < idx)) { v = rv[w2]; idx = ri[w2]; }
      sx[j] = dB[idx * 3]; sy[j] = dB[idx * 3 + 1]; sz[j] = dB[idx * 3 + 2];
      dist[idx] = -1e38f;
    }
    __syncthreads();
  }
  if (tid < Kx2) score[tid] = fabsf(sz[tid] - sz[(tid + Kx2 - 1) & (Kx2 - 1)]);
  __syncthreads();
  if (tid == 0) {                            // top-32 of score
    for (int j = 0; j < Kn; ++j) {
      float v = -1e30f; int idx = 0;
      for (int t = 0; t < Kx2; ++t)
        if (score[t] > v) { v = score[t]; idx = t; }
      stki[j] = idx; score[idx] = -1e30f;
    }
  }
  __syncthreads();
  if (tid < Kn) sw[tid] = fabsf(sz[stki[tid]] - cz);
  __syncthreads();
  if (tid == 0) {                            // softmax over K
    float mx = sw[0];
    for (int j = 1; j < Kn; ++j) mx = fmaxf(mx, sw[j]);
    float sum = 0.f;
    for (int j = 0; j < Kn; ++j) { float e = expf(sw[j] - mx); sw[j] = e; sum += e; }
    s_inv = 1.0f / sum;
  }
  __syncthreads();
  if (tid < Kn) {
    int j = tid, t = stki[j];
    float z = sw[j] * s_inv;
    size_t o = (size_t)row * Kn + j;
    snw[o * 3 + 0] = (sx[t] - cx) * z;
    snw[o * 3 + 1] = (sy[t] - cy) * z;
    snw[o * 3 + 2] = (sz[t] - cz) * z;
    tki_out[o] = t;                          // index into [0,64) — matches reference gather
    zw_out[o] = z;
  }
}

// ---------------- 3) Fused x-build + layer-1 WMMA GEMM (2 rows / block) ------
__global__ void gemm1_kernel(const float* __restrict__ feat, const float* __restrict__ snw,
                             const int* __restrict__ tki, const float* __restrict__ zw,
                             const float* __restrict__ W0, const float* __restrict__ b0,
                             _Float16* __restrict__ y1,
                             float* __restrict__ sumO, float* __restrict__ sumsqO) {
  __shared__ __align__(32) _Float16 As[4 * 3 * 512];   // 4 Mtiles x 3 Ktiles, 12KB
  __shared__ __align__(32) _Float16 Bs[3 * 4 * 512];   // 3 Ktiles x 4 Ntiles, 12KB
  __shared__ float Cs[64][H1 + 1];                     // 16.6KB
  int tid = threadIdx.x;                               // 128 threads, 4 waves
  int row0 = blockIdx.x * 2;                           // two (b,s) rows -> M=64
  for (int i = tid; i < PK1 * H1; i += 128) {
    int c = i / H1, o = i % H1;                        // c = k (0..95), o = out ch
    _Float16 w = (c < CIN) ? (_Float16)W0[o * CIN + c] : (_Float16)0.0f;
    Bs[((c >> 5) * 4 + (o >> 4)) * 512 + frag_off(o & 15, c & 31)] = w;
  }
  for (int i = tid; i < 64 * PK1; i += 128) {
    int r = i / PK1, col = i % PK1;
    int rr = row0 + (r >> 5), j = r & 31;
    size_t ro = (size_t)rr * Kn + j;
    float v;
    if (col < 3)        v = snw[ro * 3 + col];
    else if (col < CIN) {
      int bb = rr >> 10;
      int pi = tki[ro];                                // [0,64): reference feature gather
      v = feat[((size_t)bb * Np + pi) * Dn + (col - 3)] * zw[ro];
    } else v = 0.0f;
    As[((r >> 4) * 3 + (col >> 5)) * 512 + frag_off(r & 15, col & 31)] = (_Float16)v;
  }
  __syncthreads();
  int wv = tid >> 5, lane = tid & 31;
  v8f zero = {0.f, 0.f, 0.f, 0.f, 0.f, 0.f, 0.f, 0.f};
  v8f acc[4] = {zero, zero, zero, zero};               // wave wv: Mtile wv, 4 Ntiles
#pragma unroll
  for (int kt = 0; kt < 3; ++kt) {
    v16h a = load_frag(&As[(wv * 3 + kt) * 512], lane);
#pragma unroll
    for (int nt = 0; nt < 4; ++nt) {
      v16h bf = load_frag(&Bs[(kt * 4 + nt) * 512], lane);
      acc[nt] = __builtin_amdgcn_wmma_f32_16x16x32_f16(false, a, false, bf, (short)0, acc[nt], false, false);
    }
  }
  int cn = lane & 15, mh = (lane >> 4) * 8;
#pragma unroll
  for (int nt = 0; nt < 4; ++nt)
#pragma unroll
    for (int r = 0; r < 8; ++r)
      Cs[16 * wv + mh + r][16 * nt + cn] = acc[nt][r];
  __syncthreads();
  if (tid < H1) {                                      // bias + stats + f16 store
    int o = tid;
    float bias = b0[o], s = 0.f, ss = 0.f;
    for (int r = 0; r < 64; ++r) {
      float v = Cs[r][o] + bias;
      s += v; ss += v * v;
      int rr = row0 + (r >> 5), j = r & 31;
      y1[((size_t)rr * Kn + j) * H1 + o] = (_Float16)v;
    }
    atomicAdd(&sumO[o], s);
    atomicAdd(&sumsqO[o], ss);
  }
}

// ---------------- 4) BN stats -> per-channel scale/shift ---------------------
__global__ void finalize_kernel(const float* __restrict__ sum, const float* __restrict__ sumsq,
                                const float* __restrict__ g, const float* __restrict__ be,
                                float* __restrict__ scale, float* __restrict__ shift,
                                int nch, float invM) {
  int o = threadIdx.x;
  if (o < nch) {
    float m  = sum[o] * invM;
    float vr = sumsq[o] * invM - m * m;
    float sc = g[o] * rsqrtf(vr + EPSN);
    scale[o] = sc;
    shift[o] = be[o] - m * sc;
  }
}

// ---------------- 5) Mid GEMM (templated N): norm+relu -> WMMA -> bias+stats -
template <int NOUT>
__global__ void gemm_mid_kernel(const _Float16* __restrict__ yin,
                                const float* __restrict__ scIn, const float* __restrict__ shIn,
                                const float* __restrict__ W, const float* __restrict__ bias,
                                _Float16* __restrict__ yout,
                                float* __restrict__ sumO, float* __restrict__ sumsqO) {
  constexpr int NT  = NOUT / 16;     // N tiles
  constexpr int NTW = NT / 2;        // N tiles per wave (8 waves, 4 M tiles)
  __shared__ __align__(32) _Float16 As[4 * 2 * 512];        // 8KB
  __shared__ __align__(32) _Float16 Bs[2 * NT * 512];       // 8/16KB
  __shared__ float Cs[64][NOUT + 1];
  int tid = threadIdx.x;                                    // 256 threads
  size_t row0 = (size_t)blockIdx.x * 64;
  for (int i = tid; i < 64 * 64; i += 256) {
    int r = i >> 6, c = i & 63;
    float v = (float)yin[(row0 + r) * 64 + c];
    As[((r >> 4) * 2 + (c >> 5)) * 512 + frag_off(r & 15, c & 31)] =
        (_Float16)fmaxf(v * scIn[c] + shIn[c], 0.0f);
  }
  for (int i = tid; i < 64 * NOUT; i += 256) {
    int k = i / NOUT, n = i % NOUT;
    Bs[((k >> 5) * NT + (n >> 4)) * 512 + frag_off(n & 15, k & 31)] = (_Float16)W[n * 64 + k];
  }
  __syncthreads();
  int wv = tid >> 5, lane = tid & 31;
  int mt = wv & 3, half = wv >> 2;
  v8f zero = {0.f, 0.f, 0.f, 0.f, 0.f, 0.f, 0.f, 0.f};
  v8f acc[NTW];
#pragma unroll
  for (int i = 0; i < NTW; ++i) acc[i] = zero;
#pragma unroll
  for (int kt = 0; kt < 2; ++kt) {
    v16h a = load_frag(&As[(mt * 2 + kt) * 512], lane);
#pragma unroll
    for (int i = 0; i < NTW; ++i) {
      int nt = half + 2 * i;
      v16h bf = load_frag(&Bs[(kt * NT + nt) * 512], lane);
      acc[i] = __builtin_amdgcn_wmma_f32_16x16x32_f16(false, a, false, bf, (short)0, acc[i], false, false);
    }
  }
  int cn = lane & 15, mh = (lane >> 4) * 8;
#pragma unroll
  for (int i = 0; i < NTW; ++i) {
    int nt = half + 2 * i;
#pragma unroll
    for (int r = 0; r < 8; ++r)
      Cs[16 * mt + mh + r][16 * nt + cn] = acc[i][r];
  }
  __syncthreads();
  if (tid < NOUT) {
    int o = tid;
    float bv = bias[o], s = 0.f, ss = 0.f;
    for (int r = 0; r < 64; ++r) {
      float v = Cs[r][o] + bv;
      s += v; ss += v * v;
      if (yout) yout[(row0 + r) * 64 + o] = (_Float16)v;   // only when NOUT==64
    }
    atomicAdd(&sumO[o], s);
    atomicAdd(&sumsqO[o], ss);
  }
}

// ---------------- 6) Layer-3 recompute + norm + relu + max over K ------------
__global__ void final_kernel(const _Float16* __restrict__ y2,
                             const float* __restrict__ sc2, const float* __restrict__ sh2,
                             const float* __restrict__ W2, const float* __restrict__ b2,
                             const float* __restrict__ sc3, const float* __restrict__ sh3,
                             float* __restrict__ out) {
  __shared__ __align__(32) _Float16 As[2 * 2 * 512];        // 4KB
  __shared__ __align__(32) _Float16 Bs[2 * 8 * 512];        // 16KB
  __shared__ float Cs[32][129];                             // 16.5KB
  int tid = threadIdx.x;                                    // 128 threads, 4 waves
  int rowbs = blockIdx.x;                                   // (b,s)
  size_t row0 = (size_t)rowbs * Kn;
  for (int i = tid; i < 32 * 64; i += 128) {
    int r = i >> 6, c = i & 63;
    float v = (float)y2[(row0 + r) * 64 + c];
    As[((r >> 4) * 2 + (c >> 5)) * 512 + frag_off(r & 15, c & 31)] =
        (_Float16)fmaxf(v * sc2[c] + sh2[c], 0.0f);
  }
  for (int i = tid; i < 64 * 128; i += 128) {
    int k = i >> 7, n = i & 127;
    Bs[((k >> 5) * 8 + (n >> 4)) * 512 + frag_off(n & 15, k & 31)] = (_Float16)W2[n * 64 + k];
  }
  __syncthreads();
  int wv = tid >> 5, lane = tid & 31;
  int mt = wv & 1, half = wv >> 1;
  v8f zero = {0.f, 0.f, 0.f, 0.f, 0.f, 0.f, 0.f, 0.f};
  v8f acc[4] = {zero, zero, zero, zero};
#pragma unroll
  for (int kt = 0; kt < 2; ++kt) {
    v16h a = load_frag(&As[(mt * 2 + kt) * 512], lane);
#pragma unroll
    for (int i = 0; i < 4; ++i) {
      int nt = half + 2 * i;
      v16h bf = load_frag(&Bs[(kt * 8 + nt) * 512], lane);
      acc[i] = __builtin_amdgcn_wmma_f32_16x16x32_f16(false, a, false, bf, (short)0, acc[i], false, false);
    }
  }
  int cn = lane & 15, mh = (lane >> 4) * 8;
#pragma unroll
  for (int i = 0; i < 4; ++i) {
    int nt = half + 2 * i;
#pragma unroll
    for (int r = 0; r < 8; ++r)
      Cs[16 * mt + mh + r][16 * nt + cn] = acc[i][r];
  }
  __syncthreads();
  if (tid < 128) {
    int o = tid;
    float bv = b2[o], sc = sc3[o], sh = sh3[o];
    float mx = -1e30f;
    for (int r = 0; r < 32; ++r) {
      float v = fmaxf((Cs[r][o] + bv) * sc + sh, 0.0f);
      mx = fmaxf(mx, v);
    }
    out[(size_t)rowbs * 128 + o] = mx;
  }
}

// ---------------- launcher ---------------------------------------------------
extern "C" void kernel_launch(void* const* d_in, const int* in_sizes, int n_in,
                              void* d_out, int out_size, void* d_ws, size_t ws_size,
                              hipStream_t stream) {
  (void)in_sizes; (void)n_in; (void)out_size; (void)ws_size;
  const float* data  = (const float*)d_in[0];
  const float* feat  = (const float*)d_in[1];
  const int*   initf = (const int*)d_in[2];
  const float *W0 = (const float*)d_in[3],  *b0 = (const float*)d_in[4];
  const float *g0 = (const float*)d_in[5],  *be0 = (const float*)d_in[6];
  const float *W1 = (const float*)d_in[7],  *b1 = (const float*)d_in[8];
  const float *g1 = (const float*)d_in[9],  *be1 = (const float*)d_in[10];
  const float *W2 = (const float*)d_in[11], *b2 = (const float*)d_in[12];
  const float *g2 = (const float*)d_in[13], *be2 = (const float*)d_in[14];

  float* out_cent = (float*)d_out;                         // (B,S,3)
  float* out_feat = out_cent + (size_t)ROWS * 3;           // (B,S,128)

  size_t off = 0;
  auto take = [&](size_t bytes) -> void* {
    void* p = (void*)((char*)d_ws + off);
    off += (bytes + 255) & ~(size_t)255;
    return p;
  };
  float*    cent  = (float*)take((size_t)ROWS * 3 * sizeof(float));
  float*    snw   = (float*)take(MT * 3 * sizeof(float));
  int*      tki   = (int*)take(MT * sizeof(int));
  float*    zw    = (float*)take(MT * sizeof(float));
  float*    stats = (float*)take(3 * 512 * sizeof(float)); // [layer][sum|sumsq|scale|shift][128]
  _Float16* y1    = (_Float16*)take(MT * H1 * sizeof(_Float16));
  _Float16* y2    = (_Float16*)take(MT * H2 * sizeof(_Float16));

  hipMemsetAsync(stats, 0, 3 * 512 * sizeof(float), stream);

  float invM = 1.0f / (float)MT;
  fps_kernel<<<Bn, 1024, 0, stream>>>(data, initf, cent, out_cent);
  neighbor_kernel<<<ROWS, 256, 0, stream>>>(data, cent, snw, tki, zw);
  gemm1_kernel<<<ROWS / 2, 128, 0, stream>>>(feat, snw, tki, zw, W0, b0, y1,
                                             stats + 0, stats + 128);
  finalize_kernel<<<1, 128, 0, stream>>>(stats + 0, stats + 128, g0, be0,
                                         stats + 256, stats + 384, H1, invM);
  gemm_mid_kernel<H2><<<(int)(MT / 64), 256, 0, stream>>>(y1, stats + 256, stats + 384,
                                                          W1, b1, y2,
                                                          stats + 512, stats + 640);
  finalize_kernel<<<1, 128, 0, stream>>>(stats + 512, stats + 640, g1, be1,
                                         stats + 768, stats + 896, H2, invM);
  gemm_mid_kernel<H3><<<(int)(MT / 64), 256, 0, stream>>>(y2, stats + 768, stats + 896,
                                                          W2, b2, nullptr,
                                                          stats + 1024, stats + 1152);
  finalize_kernel<<<1, 128, 0, stream>>>(stats + 1024, stats + 1152, g2, be2,
                                         stats + 1280, stats + 1408, H3, invM);
  final_kernel<<<ROWS, 128, 0, stream>>>(y2, stats + 768, stats + 896, W2, b2,
                                         stats + 1280, stats + 1408, out_feat);
}
